// HeteGAT_multi_85624468013238
// MI455X (gfx1250) — matrix operand order repaired and checked by reading, not compiled, not run on hardware
//
#include <hip/hip_runtime.h>
#include <hip/hip_bf16.h>
#include <math.h>
#include <stdint.h>

// ---------------------------------------------------------------------------
// HeteGAT_multi for MI455X (gfx1250, wave32, WMMA).
// Heavy path = reading bias masks (302 MB) once via global_load_async_to_lds,
// fused masked-softmax + att@wh.  Dense MLP chain uses v_wmma_f32_16x16x32_f16
// with a transposed-B layout so both fragments load as b128.
// The 8 identical GAT heads are collapsed algebraically (tile structure).
// ---------------------------------------------------------------------------

#define NDIM   3072   // ND == NT
#define GR     4      // G*R
#define GG     2      // G

typedef __attribute__((ext_vector_type(16))) _Float16 v16h;
typedef __attribute__((ext_vector_type(8)))  float    v8f;

// ---------------------------------------------------------------------------
// Fragment loaders (f32 memory -> f16 fragments, ISA 7.12.2 layouts)
// ---------------------------------------------------------------------------
__device__ __forceinline__ v16h load_a_frag(const float* __restrict__ ap) {
  // lane(hi=0): K +0..7 | +16..23   lane(hi=1): K +8..15 | +24..31
  const float4 x0 = *reinterpret_cast<const float4*>(ap);
  const float4 x1 = *reinterpret_cast<const float4*>(ap + 4);
  const float4 x2 = *reinterpret_cast<const float4*>(ap + 16);
  const float4 x3 = *reinterpret_cast<const float4*>(ap + 20);
  const float f[16] = {x0.x, x0.y, x0.z, x0.w, x1.x, x1.y, x1.z, x1.w,
                       x2.x, x2.y, x2.z, x2.w, x3.x, x3.y, x3.z, x3.w};
  v16h a;
#pragma unroll
  for (int i = 0; i < 16; ++i) a[i] = (_Float16)f[i];
  return a;
}

__device__ __forceinline__ v16h load_b_frag(const float* __restrict__ bp) {
  // column-major W (Wt[n*K+k]); lane holds one column, K +0..7 | +16..23 (same as A)
  const float4 x0 = *reinterpret_cast<const float4*>(bp);
  const float4 x1 = *reinterpret_cast<const float4*>(bp + 4);
  const float4 x2 = *reinterpret_cast<const float4*>(bp + 16);
  const float4 x3 = *reinterpret_cast<const float4*>(bp + 20);
  const float f[16] = {x0.x, x0.y, x0.z, x0.w, x1.x, x1.y, x1.z, x1.w,
                       x2.x, x2.y, x2.z, x2.w, x3.x, x3.y, x3.z, x3.w};
  v16h b;
#pragma unroll
  for (int i = 0; i < 16; ++i) b[i] = (_Float16)f[i];
  return b;
}

// ---------------------------------------------------------------------------
// GEMM: C[M,N] = act(A[M,K] @ W[K,N] + bias[N]) with W given TRANSPOSED
// (Wt[N,K]).  One wave per 16x16 C tile, 4 waves per block along M.
// K mult of 64; M mult of 16; N mult of 16. act: 0=none 1=relu
// ---------------------------------------------------------------------------
__global__ void gemm_wmma_f16(const float* __restrict__ A,
                              const float* __restrict__ Wt,
                              const float* __restrict__ bias,
                              float* __restrict__ C,
                              int M, int N, int K, int act) {
  const int lane = threadIdx.x & 31;
  const int wave = threadIdx.x >> 5;
  const int n0   = blockIdx.x * 16;
  const int m0   = (blockIdx.y * 4 + wave) * 16;
  if (m0 >= M) return;

  const int l15 = lane & 15;
  const int hi  = lane >> 4;
  const int arow = m0 + l15;
  const int bcol = n0 + l15;

  v8f c = {0.f, 0.f, 0.f, 0.f, 0.f, 0.f, 0.f, 0.f};

  const float* ap = A  + (size_t)arow * K + hi * 8;
  const float* bp = Wt + (size_t)bcol * K + hi * 8;

  for (int kb = 0; kb < K; kb += 64) {
    v16h a0 = load_a_frag(ap + kb);
    v16h b0 = load_b_frag(bp + kb);
    c = __builtin_amdgcn_wmma_f32_16x16x32_f16(false, a0, false, b0,
                                               (short)0, c, false, false);
    v16h a1 = load_a_frag(ap + kb + 32);
    v16h b1 = load_b_frag(bp + kb + 32);
    c = __builtin_amdgcn_wmma_f32_16x16x32_f16(false, a1, false, b1,
                                               (short)0, c, false, false);
  }

  const float bv = bias ? bias[bcol] : 0.0f;
#pragma unroll
  for (int v = 0; v < 8; ++v) {
    const int row = m0 + hi * 8 + v;          // C layout: M = v + 8*hi
    float x = c[v] + bv;
    if (act == 1) x = x > 0.f ? x : 0.f;
    C[(size_t)row * N + bcol] = x;
  }
}

// One-time small-matrix transpose: Wt[n*K+k] = W[k*N+n]
__global__ void transpose_kernel(const float* __restrict__ W,
                                 float* __restrict__ Wt, int K, int N) {
  const int idx = blockIdx.x * 256 + threadIdx.x;
  if (idx >= K * N) return;
  const int k = idx / N, n = idx % N;
  Wt[(size_t)n * K + k] = W[idx];
}

// ---------------------------------------------------------------------------
// s1 = wh @ gat_a1, s2 = wh @ gat_a2   (wh: [N,16])
// ---------------------------------------------------------------------------
__global__ void gat_scores(const float* __restrict__ wh,
                           const float* __restrict__ a1,
                           const float* __restrict__ a2,
                           float* __restrict__ s1, float* __restrict__ s2, int N) {
  const int n = blockIdx.x * blockDim.x + threadIdx.x;
  if (n >= N) return;
  float x1 = 0.f, x2 = 0.f;
#pragma unroll
  for (int i = 0; i < 16; ++i) {
    const float w = wh[(size_t)n * 16 + i];
    x1 += w * a1[i];
    x2 += w * a2[i];
  }
  s1[n] = x1;
  s2[n] = x2;
}

// ---------------------------------------------------------------------------
// Fused masked-softmax + att@wh.  One block (256 thr) per (gr, row).
// The 12KB bias row is streamed into LDS with global_load_async_to_lds_b128
// (ASYNCcnt-tracked), then logits/softmax/accumulate operate out of LDS.
// head[gr,row,0:16] = elu(softmax(lrelu(s1+s2^T)+bias) @ wh)
// ---------------------------------------------------------------------------
__global__ void gat_attention(const float* __restrict__ bias,
                              const float* __restrict__ s1,
                              const float* __restrict__ s2,
                              const float* __restrict__ wh,
                              float* __restrict__ head, int N) {
  __shared__ __align__(16) float xs[NDIM];
  __shared__ float red[256];
  __shared__ float acc2[16][17];
  const int row = blockIdx.x;
  const int gr  = blockIdx.y;
  const int tid = threadIdx.x;
  const float* brow = bias + ((size_t)gr * N + row) * (size_t)N;
  const float s1r = s1[row];

  // ---- async copy: bias row -> xs (3 x b128 per thread, all lanes active)
  for (int j4 = tid * 4; j4 < N; j4 += 256 * 4) {
    const uint64_t ga = (uint64_t)(uintptr_t)(brow + j4);
    // low 32 bits of the flat shared-address == wave-relative LDS byte offset
    const uint32_t ldsoff = (uint32_t)(uintptr_t)&xs[j4];
    asm volatile("global_load_async_to_lds_b128 %0, %1, off"
                 :: "v"(ldsoff), "v"(ga) : "memory");
  }
  asm volatile("s_wait_asynccnt 0x0" ::: "memory");
  __syncthreads();

  // ---- pass 1: logits + running max (in LDS, in place)
  float mx = -INFINITY;
  for (int j = tid; j < N; j += 256) {
    float e = s1r + s2[j];
    e = e > 0.f ? e : 0.2f * e;              // leaky_relu 0.2
    const float x = e + xs[j];
    xs[j] = x;
    mx = fmaxf(mx, x);
  }
  red[tid] = mx; __syncthreads();
  for (int s = 128; s > 0; s >>= 1) {
    if (tid < s) red[tid] = fmaxf(red[tid], red[tid + s]);
    __syncthreads();
  }
  mx = red[0]; __syncthreads();

  // ---- pass 2a: p = exp(x-mx) (in place), partial denominator
  float sum = 0.f;
  for (int j = tid; j < N; j += 256) {
    const float p = __expf(xs[j] - mx);
    xs[j] = p;
    sum += p;
  }
  red[tid] = sum; __syncthreads();
  for (int s = 128; s > 0; s >>= 1) {
    if (tid < s) red[tid] += red[tid + s];
    __syncthreads();
  }
  const float inv = 1.f / red[0];
  __syncthreads();

  // ---- pass 2b: acc[i] = sum_j p[j]*wh[j,i]; 16 col-groups x 16 features
  const int i16 = tid & 15, grp = tid >> 4;
  float a = 0.f;
  for (int j = grp; j < N; j += 16)
    a += xs[j] * wh[(size_t)j * 16 + i16];
  acc2[grp][i16] = a;
  __syncthreads();
  if (tid < 16) {
    float s = 0.f;
#pragma unroll
    for (int g2 = 0; g2 < 16; ++g2) s += acc2[g2][tid];
    const float h = s * inv;
    head[((size_t)gr * N + row) * 16 + tid] = h > 0.f ? h : (__expf(h) - 1.f); // elu
  }
}

// ---------------------------------------------------------------------------
// Collapse block-tiled weights (the 8 identical heads):
//   att_wc[i,j]  = sum_b att_w[16b+i, j]          (16 x 128)
//   aom16[i]     = sum_b a_omega[16b+i]           (16)
//   dw1c[i,j]    = sum_b dec_w1[16b+i (+128), j]  (32 x 64)
// ---------------------------------------------------------------------------
__global__ void collapse_weights(const float* __restrict__ att_w,
                                 const float* __restrict__ a_omega,
                                 const float* __restrict__ dec_w1,
                                 float* __restrict__ att_wc,
                                 float* __restrict__ aom16,
                                 float* __restrict__ dw1c) {
  const int tid = blockIdx.x * blockDim.x + threadIdx.x;
  if (tid < 2048) {                    // att_wc 16x128
    const int i = tid >> 7, j = tid & 127;
    float s = 0.f;
    for (int b = 0; b < 8; ++b) s += att_w[(b * 16 + i) * 128 + j];
    att_wc[tid] = s;
  }
  if (tid < 16) {
    float s = 0.f;
    for (int b = 0; b < 8; ++b) s += a_omega[b * 16 + tid];
    aom16[tid] = s;
  }
  if (tid < 2048) {                    // dw1c 32x64
    const int i = tid >> 6, j = tid & 63;
    const int base = (i < 16) ? i : (128 + (i - 16));
    float s = 0.f;
    for (int b = 0; b < 8; ++b) s += dec_w1[(base + b * 16) * 64 + j];
    dw1c[tid] = s;
  }
}

// ---------------------------------------------------------------------------
// Semantic (meta-path) attention per (g,n) on the 16-dim tile base.
// ---------------------------------------------------------------------------
__global__ void semantic_attention(const float* __restrict__ head,
                                   const float* __restrict__ att_wc,
                                   const float* __restrict__ att_b,
                                   const float* __restrict__ att_u,
                                   float* __restrict__ pg16, int N) {
  __shared__ float hs[2][16];
  __shared__ float red[128];
  __shared__ float sc[2];
  const int n = blockIdx.x, g = blockIdx.y, tid = threadIdx.x;

  if (tid < 32) {
    const int r = tid >> 4, i = tid & 15;
    hs[r][i] = head[(((size_t)g * 2 + r) * N + n) * 16 + i];
  }
  __syncthreads();

  for (int r = 0; r < 2; ++r) {
    float v = att_b[tid];
#pragma unroll
    for (int i = 0; i < 16; ++i) v += hs[r][i] * att_wc[i * 128 + tid];
    v = tanhf(v);
    red[tid] = v * att_u[tid];
    __syncthreads();
    for (int s = 64; s > 0; s >>= 1) {
      if (tid < s) red[tid] += red[tid + s];
      __syncthreads();
    }
    if (tid == 0) sc[r] = red[0];
    __syncthreads();
  }

  const float m  = fmaxf(sc[0], sc[1]);
  const float e0 = __expf(sc[0] - m), e1 = __expf(sc[1] - m);
  const float a0 = e0 / (e0 + e1),    a1 = e1 / (e0 + e1);
  if (tid < 16)
    pg16[((size_t)g * N + n) * 16 + tid] = a0 * hs[0][tid] + a1 * hs[1][tid];
}

// gsum[g] = sum_n dot(pg16[g,n,:], aom16)   (deterministic block reduction)
__global__ void graph_accum(const float* __restrict__ pg16,
                            const float* __restrict__ aom16,
                            float* __restrict__ gsum, int N) {
  __shared__ float red[256];
  const int g = blockIdx.x, tid = threadIdx.x;
  float s = 0.f;
  for (int n = tid; n < N; n += 256) {
    const float* p = pg16 + ((size_t)g * N + n) * 16;
    float d = 0.f;
#pragma unroll
    for (int i = 0; i < 16; ++i) d += p[i] * aom16[i];
    s += d;
  }
  red[tid] = s; __syncthreads();
  for (int st = 128; st > 0; st >>= 1) {
    if (tid < st) red[tid] += red[tid + st];
    __syncthreads();
  }
  if (tid == 0) gsum[g] = red[0];
}

__global__ void graph_softmax(const float* __restrict__ gsum,
                              float* __restrict__ wsm, int N) {
  if (threadIdx.x == 0 && blockIdx.x == 0) {
    const float m0 = gsum[0] / (float)N, m1 = gsum[1] / (float)N;
    const float mx = fmaxf(m0, m1);
    const float e0 = __expf(m0 - mx), e1 = __expf(m1 - mx);
    wsm[0] = e0 / (e0 + e1);
    wsm[1] = e1 / (e0 + e1);
  }
}

__global__ void combine_graphs(const float* __restrict__ pg16,
                               const float* __restrict__ wsm,
                               float* __restrict__ final16, int N) {
  const int idx = blockIdx.x * blockDim.x + threadIdx.x;
  if (idx >= N * 16) return;
  final16[idx] = wsm[0] * pg16[idx] + wsm[1] * pg16[(size_t)N * 16 + idx];
}

// ---------------------------------------------------------------------------
// DTI decoder on edge pairs using collapsed dec_w1 (32x64).
// ---------------------------------------------------------------------------
__global__ void decoder(const int* __restrict__ ei,
                        const float* __restrict__ fd16,
                        const float* __restrict__ ft16,
                        const float* __restrict__ dw1c,
                        const float* __restrict__ dec_b1,
                        const float* __restrict__ dec_w2,
                        const float* __restrict__ dec_b2,
                        float* __restrict__ out, int E) {
  __shared__ float w1s[32 * 64];
  __shared__ float w2s[64];
  __shared__ float b1s[64];
  const int tid = threadIdx.x;
  for (int i = tid; i < 2048; i += 256) w1s[i] = dw1c[i];
  if (tid < 64) { w2s[tid] = dec_w2[tid]; b1s[tid] = dec_b1[tid]; }
  __syncthreads();

  const int e = blockIdx.x * 256 + tid;
  if (e >= E) return;
  const int d = ei[e], t = ei[E + e];
  float fd[16], ft[16];
#pragma unroll
  for (int i = 0; i < 16; ++i) {
    fd[i] = fd16[(size_t)d * 16 + i];
    ft[i] = ft16[(size_t)t * 16 + i];
  }
  float o = dec_b2[0];
  for (int j = 0; j < 64; ++j) {
    float h = b1s[j];
#pragma unroll
    for (int i = 0; i < 16; ++i)
      h += fd[i] * w1s[i * 64 + j] + ft[i] * w1s[(16 + i) * 64 + j];
    h = h > 0.f ? h : 0.f;                 // relu
    o += h * w2s[j];
  }
  out[e] = 1.f / (1.f + __expf(-o));       // sigmoid
}

// ---------------------------------------------------------------------------
extern "C" void kernel_launch(void* const* d_in, const int* in_sizes, int n_in,
                              void* d_out, int out_size, void* d_ws, size_t ws_size,
                              hipStream_t stream) {
  const int N = NDIM;
  const int E = in_sizes[4] / 2;

  const float* x_drug   = (const float*)d_in[0];
  const float* x_target = (const float*)d_in[1];
  const float* bias_d   = (const float*)d_in[2];
  const float* bias_t   = (const float*)d_in[3];
  const int*   ei       = (const int*)d_in[4];
  const float* dnn_w1 = (const float*)d_in[5];  const float* dnn_b1 = (const float*)d_in[6];
  const float* dnn_w2 = (const float*)d_in[7];  const float* dnn_b2 = (const float*)d_in[8];
  const float* dnn_w3 = (const float*)d_in[9];  const float* dnn_b3 = (const float*)d_in[10];
  const float* tnn_w1 = (const float*)d_in[11]; const float* tnn_b1 = (const float*)d_in[12];
  const float* tnn_w2 = (const float*)d_in[13]; const float* tnn_b2 = (const float*)d_in[14];
  const float* tnn_w3 = (const float*)d_in[15]; const float* tnn_b3 = (const float*)d_in[16];
  const float* gat_w  = (const float*)d_in[17];
  const float* gat_a1 = (const float*)d_in[18]; const float* gat_a2 = (const float*)d_in[19];
  const float* att_w  = (const float*)d_in[20]; const float* att_b  = (const float*)d_in[21];
  const float* att_u  = (const float*)d_in[22];
  const float* a_omega = (const float*)d_in[23];
  const float* dec_w1 = (const float*)d_in[24]; const float* dec_b1 = (const float*)d_in[25];
  const float* dec_w2 = (const float*)d_in[26]; const float* dec_b2 = (const float*)d_in[27];
  float* out = (float*)d_out;

  // ---- workspace layout (floats, 16B aligned chunks) ----
  float* ws = (float*)d_ws;
  size_t off = 0;
  auto alloc = [&](size_t nf) { float* p = ws + off; off += (nf + 3) & ~(size_t)3; return p; };
  float* final_d16 = alloc((size_t)N * 16);
  float* final_t16 = alloc((size_t)N * 16);
  float* att_wc    = alloc(16 * 128);
  float* aom16     = alloc(16);
  float* dw1c      = alloc(32 * 64);
  float* gsum      = alloc(4);
  float* wsm       = alloc(4);
  float* wt        = alloc((size_t)1280 * 512);   // transposed-weight scratch
  // per-branch scratch (reused drug -> target)
  float* h1    = alloc((size_t)N * 512);
  float* h2    = alloc((size_t)N * 256);
  float* feats = alloc((size_t)N * 128);
  float* wh    = alloc((size_t)N * 16);
  float* s1    = alloc(N);
  float* s2    = alloc(N);
  float* headb = alloc((size_t)GR * N * 16);
  float* pg16  = alloc((size_t)GG * N * 16);
  (void)ws_size; (void)n_in; (void)out_size;

  collapse_weights<<<8, 256, 0, stream>>>(att_w, a_omega, dec_w1, att_wc, aom16, dw1c);

  auto run_branch = [&](const float* x, int Kin,
                        const float* w1, const float* b1,
                        const float* w2, const float* b2,
                        const float* w3, const float* b3,
                        const float* bias, float* final16) {
    const dim3 blk(128);
    transpose_kernel<<<(Kin * 512 + 255) / 256, 256, 0, stream>>>(w1, wt, Kin, 512);
    gemm_wmma_f16<<<dim3(512 / 16, N / 64), blk, 0, stream>>>(x,  wt, b1, h1,    N, 512, Kin, 1);
    transpose_kernel<<<(512 * 256 + 255) / 256, 256, 0, stream>>>(w2, wt, 512, 256);
    gemm_wmma_f16<<<dim3(256 / 16, N / 64), blk, 0, stream>>>(h1, wt, b2, h2,    N, 256, 512, 1);
    transpose_kernel<<<(256 * 128 + 255) / 256, 256, 0, stream>>>(w3, wt, 256, 128);
    gemm_wmma_f16<<<dim3(128 / 16, N / 64), blk, 0, stream>>>(h2, wt, b3, feats, N, 128, 256, 1);
    transpose_kernel<<<(128 * 16 + 255) / 256, 256, 0, stream>>>(gat_w, wt, 128, 16);
    gemm_wmma_f16<<<dim3(16 / 16,  N / 64), blk, 0, stream>>>(feats, wt, nullptr, wh, N, 16, 128, 0);
    gat_scores<<<(N + 255) / 256, 256, 0, stream>>>(wh, gat_a1, gat_a2, s1, s2, N);
    gat_attention<<<dim3(N, GR), 256, 0, stream>>>(bias, s1, s2, wh, headb, N);
    semantic_attention<<<dim3(N, GG), 128, 0, stream>>>(headb, att_wc, att_b, att_u, pg16, N);
    graph_accum<<<GG, 256, 0, stream>>>(pg16, aom16, gsum, N);
    graph_softmax<<<1, 32, 0, stream>>>(gsum, wsm, N);
    combine_graphs<<<(N * 16 + 255) / 256, 256, 0, stream>>>(pg16, wsm, final16, N);
  };

  run_branch(x_drug,   1024, dnn_w1, dnn_b1, dnn_w2, dnn_b2, dnn_w3, dnn_b3, bias_d, final_d16);
  run_branch(x_target, 1280, tnn_w1, tnn_b1, tnn_w2, tnn_b2, tnn_w3, tnn_b3, bias_t, final_t16);

  decoder<<<(E + 255) / 256, 256, 0, stream>>>(ei, final_d16, final_t16, dw1c,
                                               dec_b1, dec_w2, dec_b2, out, E);
}
